// EncoderGrp_30382598652302
// MI455X (gfx1250) — matrix operand order repaired and checked by reading, not compiled, not run on hardware
//
#include <hip/hip_runtime.h>
#include <hip/hip_bf16.h>
#include <stdint.h>

// Problem constants (match reference)
#define B_    8
#define N_    8192
#define C_IN_ 64
#define DIM_  128
#define M_    2048
#define K_    32

typedef __attribute__((ext_vector_type(16))) __bf16 v16bf;
typedef __attribute__((ext_vector_type(8)))  float  v8f;
typedef __attribute__((ext_vector_type(4)))  unsigned int v4u;

// ---------------------------------------------------------------------------
// Kernel 0: transpose + bf16-convert weights.
// W1 [67,128] -> W1T [128][96] (cols 67..95 zero), W2 [128,128] -> W2T [128][128]
// ---------------------------------------------------------------------------
__global__ void prep_weights_kernel(const float* __restrict__ W1,
                                    const float* __restrict__ W2,
                                    __bf16* __restrict__ W1T,
                                    __bf16* __restrict__ W2T) {
    int stride = gridDim.x * blockDim.x;
    int t = blockIdx.x * blockDim.x + threadIdx.x;
    for (int i = t; i < 128 * 96; i += stride) {
        int n = i / 96, k = i % 96;
        W1T[i] = (k < 67) ? (__bf16)W1[k * 128 + n] : (__bf16)0.0f;
    }
    for (int i = t; i < 128 * 128; i += stride) {
        int n = i >> 7, k = i & 127;
        W2T[i] = (__bf16)W2[k * 128 + n];
    }
}

// ---------------------------------------------------------------------------
// Kernel 1: farthest point sampling. One block per batch, 1024 threads,
// 8 points per thread held in registers. Writes pc_fps output directly.
// ---------------------------------------------------------------------------
__global__ __launch_bounds__(1024)
void fps_kernel(const float* __restrict__ pc, float* __restrict__ pcfps) {
    int b = blockIdx.x;
    int t = threadIdx.x;
    const float* pcb = pc + (size_t)b * N_ * 3;

    float px[8], py[8], pz[8], dist[8];
#pragma unroll
    for (int j = 0; j < 8; ++j) {
        int n = t + j * 1024;
        px[j] = pcb[n * 3 + 0];
        py[j] = pcb[n * 3 + 1];
        pz[j] = pcb[n * 3 + 2];
        dist[j] = 1e10f;
    }

    __shared__ float curS[3];
    __shared__ float redV[32];
    __shared__ int   redI[32];

    if (t == 0) {
        curS[0] = pcb[0]; curS[1] = pcb[1]; curS[2] = pcb[2];
        float* o = pcfps + (size_t)b * M_ * 3;
        o[0] = pcb[0]; o[1] = pcb[1]; o[2] = pcb[2];
    }
    __syncthreads();

    for (int s = 1; s < M_; ++s) {
        float cx = curS[0], cy = curS[1], cz = curS[2];
        float bv = -1.0f; int bidx = 0x7fffffff;
#pragma unroll
        for (int j = 0; j < 8; ++j) {
            float dx = px[j] - cx, dy = py[j] - cy, dz = pz[j] - cz;
            float d = dx * dx + dy * dy + dz * dz;
            d = fminf(d, dist[j]);
            dist[j] = d;
            int n = t + j * 1024;
            bool better = (d > bv) || (d == bv && n < bidx);
            bv = better ? d : bv;
            bidx = better ? n : bidx;
        }
        // wave32 argmax reduce (tie -> lowest index, matches jnp.argmax)
#pragma unroll
        for (int off = 16; off > 0; off >>= 1) {
            float ov = __shfl_down(bv, off, 32);
            int   oi = __shfl_down(bidx, off, 32);
            bool better = (ov > bv) || (ov == bv && oi < bidx);
            bv = better ? ov : bv;
            bidx = better ? oi : bidx;
        }
        if ((t & 31) == 0) { redV[t >> 5] = bv; redI[t >> 5] = bidx; }
        __syncthreads();
        if (t < 32) {
            bv = redV[t]; bidx = redI[t];
#pragma unroll
            for (int off = 16; off > 0; off >>= 1) {
                float ov = __shfl_down(bv, off, 32);
                int   oi = __shfl_down(bidx, off, 32);
                bool better = (ov > bv) || (ov == bv && oi < bidx);
                bv = better ? ov : bv;
                bidx = better ? oi : bidx;
            }
            if (t == 0) {
                float qx = pcb[bidx * 3 + 0];
                float qy = pcb[bidx * 3 + 1];
                float qz = pcb[bidx * 3 + 2];
                curS[0] = qx; curS[1] = qy; curS[2] = qz;
                float* o = pcfps + ((size_t)b * M_ + s) * 3;
                o[0] = qx; o[1] = qy; o[2] = qz;
            }
        }
        __syncthreads();
    }
}

// ---------------------------------------------------------------------------
// Kernel 2: KNN (32 nearest, ascending). One thread per query; pc staged in
// LDS tiles; fully unrolled select-based insertion (no scratch spills).
// ---------------------------------------------------------------------------
#define KNN_TILE 512
__global__ __launch_bounds__(256)
void knn_kernel(const float* __restrict__ pc,
                const float* __restrict__ pcfps,
                int* __restrict__ knn_idx) {
    int gq = blockIdx.x * 256 + threadIdx.x;  // [0, B*M)
    int b = gq / M_;
    const float* pcb = pc + (size_t)b * N_ * 3;
    float qx = pcfps[(size_t)gq * 3 + 0];
    float qy = pcfps[(size_t)gq * 3 + 1];
    float qz = pcfps[(size_t)gq * 3 + 2];

    float bd[K_]; int bi[K_];
#pragma unroll
    for (int i = 0; i < K_; ++i) { bd[i] = 3.4e38f; bi[i] = 0; }

    __shared__ float tx[KNN_TILE], ty[KNN_TILE], tz[KNN_TILE];

    for (int tile = 0; tile < N_ / KNN_TILE; ++tile) {
        __syncthreads();
        for (int i = threadIdx.x; i < KNN_TILE; i += 256) {
            const float* p = pcb + (size_t)(tile * KNN_TILE + i) * 3;
            tx[i] = p[0]; ty[i] = p[1]; tz[i] = p[2];
        }
        __syncthreads();
        for (int i = 0; i < KNN_TILE; ++i) {
            float dx = tx[i] - qx, dy = ty[i] - qy, dz = tz[i] - qz;
            float d = dx * dx + dy * dy + dz * dz;
            if (d < bd[K_ - 1]) {
                float cd = d; int ci = tile * KNN_TILE + i;
#pragma unroll
                for (int p = K_ - 1; p >= 0; --p) {
                    bool up = (p > 0) && (cd < bd[p - 1]);   // shift from above
                    bool here = (cd < bd[p]);
                    float nd = up ? bd[p - 1] : (here ? cd : bd[p]);
                    int   ni = up ? bi[p - 1] : (here ? ci : bi[p]);
                    bd[p] = nd; bi[p] = ni;
                }
            }
        }
    }
    int* o = knn_idx + (size_t)gq * K_;
#pragma unroll
    for (int i = 0; i < K_; ++i) o[i] = bi[i];
}

// ---------------------------------------------------------------------------
// Kernel 3: gather + MLP (bf16 WMMA, f32 accumulate) + max-pool over K.
// Block = 256 threads (8 waves) handles 4 query points = 128 rows.
// A-fragments (row tiles, per wave) hoisted out of the column-tile loop so
// LDS reads happen once per k-step, not once per (ct, k-step).
// ---------------------------------------------------------------------------
#define BQ 4
#define ROWS 128          // BQ * K_
#define KPAD1 96          // 67 padded to 3*32

__global__ __launch_bounds__(256)
void mlp_kernel(const float* __restrict__ x,
                const float* __restrict__ pc,
                const float* __restrict__ b1,
                const float* __restrict__ b2,
                const __bf16* __restrict__ W1T,
                const __bf16* __restrict__ W2T,
                const int* __restrict__ knn_idx,
                const float* __restrict__ pcfps,
                float* __restrict__ out) {
    __shared__ __attribute__((aligned(16))) __bf16 featS[ROWS * KPAD1]; // 24 KB
    __shared__ __attribute__((aligned(16))) __bf16 h1S[ROWS * DIM_];    // 32 KB
    __shared__ __attribute__((aligned(16))) float  pmaxS[16 * DIM_];    // 8 KB

    int blk = blockIdx.x;
    int b = blk / (M_ / BQ);
    int m0 = (blk % (M_ / BQ)) * BQ;
    int t = threadIdx.x;

    // ---- gather & convert features into LDS (2 threads per row) ----
    {
        int r = t >> 1;        // 0..127
        int half = t & 1;
        int q = r >> 5, kk = r & 31;
        int m = m0 + q;
        int idx = knn_idx[((size_t)(b * M_ + m)) * K_ + kk];
        const float* xr = x + ((size_t)b * N_ + idx) * C_IN_;
        const float* pr = pc + ((size_t)b * N_ + idx) * 3;
        const float* qf = pcfps + ((size_t)(b * M_ + m)) * 3;
        __bf16* fr = featS + r * KPAD1;
        if (half == 0) {
            fr[0] = (__bf16)(pr[0] - qf[0]);
            fr[1] = (__bf16)(pr[1] - qf[1]);
            fr[2] = (__bf16)(pr[2] - qf[2]);
            for (int c = 0; c < 45; ++c) fr[3 + c] = (__bf16)xr[c];
        } else {
            for (int c = 45; c < 64; ++c) fr[3 + c] = (__bf16)xr[c];
            for (int c = 67; c < KPAD1; ++c) fr[c] = (__bf16)0.0f;
        }
    }
    __syncthreads();

    int wave = t >> 5;       // row tile (0..7)
    int lane = t & 31;
    int lhalf = lane >> 4;   // 0/1
    int l16 = lane & 15;

    // ---- GEMM1: feat @ W1 + b1, ReLU -> h1S (bf16) ----
    {
        v16bf a1[3];
#pragma unroll
        for (int ks = 0; ks < 3; ++ks) {
            // A fragment: row = wave*16 + l16; k runs [ks*32+lhalf*8, +8) and +16
            const __bf16* ap = featS + (wave * 16 + l16) * KPAD1 + ks * 32 + lhalf * 8;
            v4u* aq = reinterpret_cast<v4u*>(&a1[ks]);
            aq[0] = *reinterpret_cast<const v4u*>(ap);
            aq[1] = *reinterpret_cast<const v4u*>(ap + 16);
        }
        for (int ct = 0; ct < 8; ++ct) {
            v8f acc = {0.f, 0.f, 0.f, 0.f, 0.f, 0.f, 0.f, 0.f};
#pragma unroll
            for (int ks = 0; ks < 3; ++ks) {
                v16bf bf;
                // B fragment: n = ct*16 + l16; k runs [ks*32+lhalf*8, +8) and +16
                const __bf16* bp = W1T + (ct * 16 + l16) * KPAD1 + ks * 32 + lhalf * 8;
                v4u* bq = reinterpret_cast<v4u*>(&bf);
                bq[0] = *reinterpret_cast<const v4u*>(bp);
                bq[1] = *reinterpret_cast<const v4u*>(bp + 16);
                acc = __builtin_amdgcn_wmma_f32_16x16x32_bf16(false, a1[ks], false, bf,
                                                              (short)0, acc, false, false);
            }
            float bias = b1[ct * 16 + l16];
#pragma unroll
            for (int r = 0; r < 8; ++r) {
                float h = acc[r] + bias;
                h = h > 0.0f ? h : 0.0f;
                h1S[(wave * 16 + r + 8 * lhalf) * DIM_ + ct * 16 + l16] = (__bf16)h;
            }
        }
    }
    __syncthreads();

    // ---- GEMM2: h1 @ W2, per-tile max over rows -> pmaxS ----
    {
        v16bf a2[4];
#pragma unroll
        for (int ks = 0; ks < 4; ++ks) {
            const __bf16* ap = h1S + (wave * 16 + l16) * DIM_ + ks * 32 + lhalf * 8;
            v4u* aq = reinterpret_cast<v4u*>(&a2[ks]);
            aq[0] = *reinterpret_cast<const v4u*>(ap);
            aq[1] = *reinterpret_cast<const v4u*>(ap + 16);
        }
        for (int ct = 0; ct < 8; ++ct) {
            v8f acc = {0.f, 0.f, 0.f, 0.f, 0.f, 0.f, 0.f, 0.f};
#pragma unroll
            for (int ks = 0; ks < 4; ++ks) {
                v16bf bf;
                const __bf16* bp = W2T + (ct * 16 + l16) * DIM_ + ks * 32 + lhalf * 8;
                v4u* bq = reinterpret_cast<v4u*>(&bf);
                bq[0] = *reinterpret_cast<const v4u*>(bp);
                bq[1] = *reinterpret_cast<const v4u*>(bp + 16);
                acc = __builtin_amdgcn_wmma_f32_16x16x32_bf16(false, a2[ks], false, bf,
                                                              (short)0, acc, false, false);
            }
            float mx = acc[0];
#pragma unroll
            for (int r = 1; r < 8; ++r) mx = fmaxf(mx, acc[r]);
            // lane holds max over 8 rows: rowhalf = wave*2 + lhalf
            pmaxS[(wave * 2 + lhalf) * DIM_ + ct * 16 + l16] = mx;
        }
    }
    __syncthreads();

    // ---- final max over 4 rowhalves per query (+ b2) ----
    for (int o = t; o < BQ * DIM_; o += 256) {
        int q = o >> 7, n = o & 127;
        float v = pmaxS[(q * 4 + 0) * DIM_ + n];
        v = fmaxf(v, pmaxS[(q * 4 + 1) * DIM_ + n]);
        v = fmaxf(v, pmaxS[(q * 4 + 2) * DIM_ + n]);
        v = fmaxf(v, pmaxS[(q * 4 + 3) * DIM_ + n]);
        out[((size_t)(b * M_ + m0 + q)) * DIM_ + n] = v + b2[n];
    }
}

// ---------------------------------------------------------------------------
extern "C" void kernel_launch(void* const* d_in, const int* in_sizes, int n_in,
                              void* d_out, int out_size, void* d_ws, size_t ws_size,
                              hipStream_t stream) {
    const float* x  = (const float*)d_in[0];
    const float* pc = (const float*)d_in[1];
    const float* W1 = (const float*)d_in[2];
    const float* b1 = (const float*)d_in[3];
    const float* W2 = (const float*)d_in[4];
    const float* b2 = (const float*)d_in[5];

    float* out   = (float*)d_out;                       // [B,M,DIM]
    float* pcfps = out + (size_t)B_ * M_ * DIM_;        // [B,M,3]

    char* ws = (char*)d_ws;
    int*    knn_idx = (int*)ws;                                        // 2 MB
    __bf16* W1T = (__bf16*)(ws + (size_t)B_ * M_ * K_ * sizeof(int));  // 24 KB
    __bf16* W2T = W1T + 128 * 96;                                      // 32 KB

    prep_weights_kernel<<<dim3(48), dim3(256), 0, stream>>>(W1, W2, W1T, W2T);
    fps_kernel<<<dim3(B_), dim3(1024), 0, stream>>>(pc, pcfps);
    knn_kernel<<<dim3((B_ * M_) / 256), dim3(256), 0, stream>>>(pc, pcfps, knn_idx);
    mlp_kernel<<<dim3((B_ * M_) / BQ), dim3(256), 0, stream>>>(
        x, pc, b1, b2, W1T, W2T, knn_idx, pcfps, out);
}
